// GAT_670014898213
// MI455X (gfx1250) — matrix operand order repaired and checked by reading, not compiled, not run on hardware
//
#include <hip/hip_runtime.h>
#include <hip/hip_bf16.h>
#include <stdint.h>

#define N_NODES   50000
#define N_EDGES   800000
#define HEADS     8
#define DIM_ENC   64
#define DIM_HID   512   /* HEADS * DIM_ENC */
#define DIM_MLP   512
#define NUM_GRAPHS 2000
#define NEG_SLOPE 0.2f

typedef __bf16 bf16_t;
typedef __attribute__((ext_vector_type(8)))  __bf16 v8bf;
typedef __attribute__((ext_vector_type(16))) __bf16 v16bf;
typedef __attribute__((ext_vector_type(8)))  float  v8f;

// ---------------------------------------------------------------- utilities

__device__ __forceinline__ bf16_t f2bf(float f) {
  unsigned u = __float_as_uint(f);
  u += 0x7FFFu + ((u >> 16) & 1u);            // round-to-nearest-even
  unsigned short h = (unsigned short)(u >> 16);
  bf16_t b;
  __builtin_memcpy(&b, &h, sizeof(h));
  return b;
}

__device__ __forceinline__ void atomicMaxF(float* addr, float val) {
  int* ia = (int*)addr;
  int old = __float_as_int(*(volatile float*)addr);
  while (__int_as_float(old) < val) {
    int prev = atomicCAS(ia, old, __float_as_int(val));
    if (prev == old) break;
    old = prev;
  }
}

__global__ void fill_f32(float* __restrict__ p, float v, int n) {
  int i = blockIdx.x * blockDim.x + threadIdx.x;
  if (i < n) p[i] = v;
}

__global__ void cvt_f32_bf16(const float* __restrict__ in, bf16_t* __restrict__ out, int n) {
  int i = blockIdx.x * blockDim.x + threadIdx.x;
  if (i < n) out[i] = f2bf(in[i]);
}

// in [R,C] row-major f32  ->  out [C,R] bf16 (i.e. B^T, so GEMM reads K contiguously)
__global__ void transpose_cvt_bf16(const float* __restrict__ in, bf16_t* __restrict__ out,
                                   int R, int C) {
  int i = blockIdx.x * blockDim.x + threadIdx.x;
  if (i >= R * C) return;
  int r = i / C, c = i - r * C;
  out[(size_t)c * R + r] = f2bf(in[i]);
}

// ---------------------------------------------------------------- WMMA GEMM
// C[M,N] = A[M,K] * B[K,N], A row-major bf16, Bt = B^T [N,K] bf16, C f32.
// One wave32 computes one 16x16 tile; K stepped by 32 via v_wmma_f32_16x16x32_bf16.
// A fragment (16-bit A 16x32 layout): lane half h, row r=lane&15:
//   K = h*8 + {0..7}  (VGPR0-3)  and  16 + h*8 + {0..7}  (VGPR4-7)   -> two 16B loads
// B fragment (16-bit B 32x16 layout): col r=lane&15, K = h*8 + {0..7} and 16 + h*8 + {0..7} -> two 16B loads
__global__ __launch_bounds__(256) void wmma_gemm_bf16(
    const bf16_t* __restrict__ A, const bf16_t* __restrict__ Bt,
    float* __restrict__ C, int M, int N, int K) {
  int gtid  = blockIdx.x * blockDim.x + threadIdx.x;
  int wave  = gtid >> 5;
  int lane  = gtid & 31;
  int tilesN = N >> 4;
  int tilesM = M >> 4;
  if (wave >= tilesM * tilesN) return;
  int tm = wave / tilesN, tn = wave - tm * tilesN;
  int m0 = tm << 4, n0 = tn << 4;
  int half = lane >> 4;
  int r    = lane & 15;

  const bf16_t* Arow = A  + (size_t)(m0 + r) * K + half * 8;
  const bf16_t* Brow = Bt + (size_t)(n0 + r) * K + half * 8;

  v8f acc = {};
  for (int k0 = 0; k0 < K; k0 += 32) {
    v8bf alo = *(const v8bf*)(Arow + k0);
    v8bf ahi = *(const v8bf*)(Arow + k0 + 16);
    v8bf blo = *(const v8bf*)(Brow + k0);
    v8bf bhi = *(const v8bf*)(Brow + k0 + 16);
    v16bf a, b;
#pragma unroll
    for (int i = 0; i < 8; ++i) {
      a[i] = alo[i]; a[i + 8] = ahi[i];
      b[i] = blo[i]; b[i + 8] = bhi[i];
    }
    acc = __builtin_amdgcn_wmma_f32_16x16x32_bf16(
        /*neg_a=*/false, a, /*neg_b=*/false, b,
        /*c_mod=*/(short)0, acc, /*reuse_a=*/false, /*reuse_b=*/false);
  }
#pragma unroll
  for (int i = 0; i < 8; ++i)
    C[(size_t)(m0 + i + half * 8) * N + n0 + r] = acc[i];
}

// ---------------------------------------------------------------- GAT pieces

// alpha_src/dst[n,h] = sum_f hfeat[n, h*64+f] * a_{src,dst}[h,f]
__global__ void node_alpha(const float* __restrict__ hfeat,
                           const float* __restrict__ a_src, const float* __restrict__ a_dst,
                           float* __restrict__ as_, float* __restrict__ ad_, int N) {
  int idx = blockIdx.x * blockDim.x + threadIdx.x;   // n*8 + h
  if (idx >= N * HEADS) return;
  int h = idx & 7;
  const float* hv = hfeat + (size_t)idx * DIM_ENC;   // [N,8,64] contiguous
  float ss = 0.f, sd = 0.f;
#pragma unroll 8
  for (int f = 0; f < DIM_ENC; ++f) {
    float v = hv[f];
    ss += v * a_src[h * DIM_ENC + f];
    sd += v * a_dst[h * DIM_ENC + f];
  }
  as_[idx] = ss;
  ad_[idx] = sd;
}

__device__ __forceinline__ void edge_sd(const int* __restrict__ ei, int e, int E,
                                        int& src, int& dst) {
  if (e < E) { src = ei[e]; dst = ei[E + e]; }
  else       { src = dst = e - E; }             // self loops appended
}

__global__ void edge_max(const int* __restrict__ ei, const float* __restrict__ as_,
                         const float* __restrict__ ad_, float* __restrict__ m_,
                         int E, int N) {
  int t = blockIdx.x * blockDim.x + threadIdx.x;
  if (t >= (E + N) * HEADS) return;
  int e = t >> 3, h = t & 7, src, dst;
  edge_sd(ei, e, E, src, dst);
  float v = as_[src * HEADS + h] + ad_[dst * HEADS + h];
  v = v > 0.f ? v : NEG_SLOPE * v;
  atomicMaxF(&m_[dst * HEADS + h], v);
}

__global__ void edge_expsum(const int* __restrict__ ei, const float* __restrict__ as_,
                            const float* __restrict__ ad_, const float* __restrict__ m_,
                            float* __restrict__ s_, int E, int N) {
  int t = blockIdx.x * blockDim.x + threadIdx.x;
  if (t >= (E + N) * HEADS) return;
  int e = t >> 3, h = t & 7, src, dst;
  edge_sd(ei, e, E, src, dst);
  float v = as_[src * HEADS + h] + ad_[dst * HEADS + h];
  v = v > 0.f ? v : NEG_SLOPE * v;
  atomicAdd(&s_[dst * HEADS + h], __expf(v - m_[dst * HEADS + h]));
}

__global__ void edge_aggregate(const int* __restrict__ ei, const float* __restrict__ as_,
                               const float* __restrict__ ad_, const float* __restrict__ m_,
                               const float* __restrict__ s_, const float* __restrict__ hfeat,
                               float* __restrict__ out, int E, int N) {
  int t = blockIdx.x * blockDim.x + threadIdx.x;
  if (t >= (E + N) * HEADS) return;
  int e = t >> 3, h = t & 7, src, dst;
  edge_sd(ei, e, E, src, dst);
  float v = as_[src * HEADS + h] + ad_[dst * HEADS + h];
  v = v > 0.f ? v : NEG_SLOPE * v;
  float ex    = __expf(v - m_[dst * HEADS + h]);
  float alpha = ex / (s_[dst * HEADS + h] + 1e-16f);
  const float* hs = hfeat + (size_t)src * DIM_HID + h * DIM_ENC;
  float*       od = out   + (size_t)dst * DIM_HID + h * DIM_ENC;
#pragma unroll 8
  for (int f = 0; f < DIM_ENC; ++f) atomicAdd(&od[f], hs[f] * alpha);
}

// concat=False: mean over heads + bias -> bf16 for the next GEMM
__global__ void mean_heads_bias(const float* __restrict__ out, const float* __restrict__ b1,
                                bf16_t* __restrict__ h1b, int N) {
  int idx = blockIdx.x * blockDim.x + threadIdx.x;   // n*64 + f
  if (idx >= N * DIM_ENC) return;
  int n = idx >> 6, f = idx & 63;
  const float* o = out + (size_t)n * DIM_HID + f;
  float s = 0.f;
#pragma unroll
  for (int h = 0; h < HEADS; ++h) s += o[h * DIM_ENC];
  h1b[idx] = f2bf(s * 0.125f + b1[f]);
}

// concat=True output + bias, pooled by graph id via atomics
__global__ void bias_pool(const float* __restrict__ out2, const float* __restrict__ b2,
                          const int* __restrict__ batch, float* __restrict__ g, int N) {
  int idx = blockIdx.x * blockDim.x + threadIdx.x;   // n*512 + c
  if (idx >= N * DIM_HID) return;
  int n = idx >> 9, c = idx & 511;
  atomicAdd(&g[(size_t)batch[n] * DIM_HID + c], out2[idx] + b2[c]);
}

// x = relu(x + b) in place, plus bf16 copy for the next GEMM
__global__ void bias_relu_cvt(float* __restrict__ x, const float* __restrict__ b,
                              bf16_t* __restrict__ xb, int rows, int cols) {
  int idx = blockIdx.x * blockDim.x + threadIdx.x;
  if (idx >= rows * cols) return;
  int c = idx % cols;
  float v = x[idx] + b[c];
  v = v > 0.f ? v : 0.f;
  x[idx]  = v;
  xb[idx] = f2bf(v);
}

// out[g] = dot(g2[g,:512], w3) + b3   (wave32 per graph)
__global__ void final_fc(const float* __restrict__ g2, const float* __restrict__ w3,
                         const float* __restrict__ b3, float* __restrict__ out, int G) {
  int gtid = blockIdx.x * blockDim.x + threadIdx.x;
  int wave = gtid >> 5, lane = gtid & 31;
  if (wave >= G) return;
  const float* row = g2 + (size_t)wave * DIM_MLP;
  float v = 0.f;
  for (int c = lane; c < DIM_MLP; c += 32) v += row[c] * w3[c];
#pragma unroll
  for (int off = 16; off; off >>= 1) v += __shfl_xor(v, off, 32);
  if (lane == 0) out[wave] = v + b3[0];
}

// ---------------------------------------------------------------- launcher

static inline int nblk(long long n, int t) { return (int)((n + t - 1) / t); }

static void run_gemm(const bf16_t* A, const bf16_t* Bt, float* C,
                     int M, int N, int K, hipStream_t s) {
  long long threads = (long long)(M / 16) * (N / 16) * 32;
  wmma_gemm_bf16<<<nblk(threads, 256), 256, 0, s>>>(A, Bt, C, M, N, K);
}

extern "C" void kernel_launch(void* const* d_in, const int* in_sizes, int n_in,
                              void* d_out, int out_size, void* d_ws, size_t ws_size,
                              hipStream_t stream) {
  (void)in_sizes; (void)n_in; (void)out_size; (void)ws_size;

  const float* x      = (const float*)d_in[0];
  const int*   ei     = (const int*)  d_in[1];
  const int*   batch  = (const int*)  d_in[2];
  const float* W1     = (const float*)d_in[3];
  const float* asrc1  = (const float*)d_in[4];
  const float* adst1  = (const float*)d_in[5];
  const float* b1     = (const float*)d_in[6];
  const float* W2     = (const float*)d_in[7];
  const float* asrc2  = (const float*)d_in[8];
  const float* adst2  = (const float*)d_in[9];
  const float* b2     = (const float*)d_in[10];
  const float* fcW1   = (const float*)d_in[11];
  const float* fcb1   = (const float*)d_in[12];
  const float* fcW2   = (const float*)d_in[13];
  const float* fcb2   = (const float*)d_in[14];
  const float* fcW3   = (const float*)d_in[15];
  const float* fcb3   = (const float*)d_in[16];
  float* out = (float*)d_out;

  // ---- workspace layout (256B aligned)
  char*  ws  = (char*)d_ws;
  size_t off = 0;
  auto alloc = [&](size_t bytes) { char* p = ws + off; off += (bytes + 255) & ~(size_t)255; return p; };
  const int N = N_NODES, E = N_EDGES, G = NUM_GRAPHS;

  float*  hfeat = (float*) alloc((size_t)N * DIM_HID * 4);   // [N,512] per-head features
  float*  outac = (float*) alloc((size_t)N * DIM_HID * 4);   // [N,512] scatter accumulator
  float*  as_   = (float*) alloc((size_t)N * HEADS * 4);
  float*  ad_   = (float*) alloc((size_t)N * HEADS * 4);
  float*  m_    = (float*) alloc((size_t)N * HEADS * 4);
  float*  s_    = (float*) alloc((size_t)N * HEADS * 4);
  bf16_t* xbf   = (bf16_t*)alloc((size_t)N * 32 * 2);
  bf16_t* wbf   = (bf16_t*)alloc((size_t)512 * 512 * 2);     // transposed weight (reused)
  bf16_t* h1bf  = (bf16_t*)alloc((size_t)N * DIM_ENC * 2);
  float*  g_    = (float*) alloc((size_t)G * DIM_HID * 4);
  bf16_t* gbf   = (bf16_t*)alloc((size_t)G * DIM_HID * 2);
  float*  g1f   = (float*) alloc((size_t)G * DIM_MLP * 4);
  bf16_t* g1bf  = (bf16_t*)alloc((size_t)G * DIM_MLP * 2);
  float*  g2f   = (float*) alloc((size_t)G * DIM_MLP * 4);

  const int TB = 256;
  const long long nEH = (long long)(E + N) * HEADS;

  // ================= Layer 1: GATConv(x, W1) concat=False =================
  cvt_f32_bf16<<<nblk((long long)N * 32, TB), TB, 0, stream>>>(x, xbf, N * 32);
  transpose_cvt_bf16<<<nblk(32 * 512, TB), TB, 0, stream>>>(W1, wbf, 32, 512);
  run_gemm(xbf, wbf, hfeat, N, DIM_HID, 32, stream);

  node_alpha<<<nblk((long long)N * HEADS, TB), TB, 0, stream>>>(hfeat, asrc1, adst1, as_, ad_, N);
  fill_f32<<<nblk((long long)N * HEADS, TB), TB, 0, stream>>>(m_, -1e30f, N * HEADS);
  fill_f32<<<nblk((long long)N * HEADS, TB), TB, 0, stream>>>(s_, 0.f, N * HEADS);
  fill_f32<<<nblk((long long)N * DIM_HID, TB), TB, 0, stream>>>(outac, 0.f, N * DIM_HID);

  edge_max     <<<nblk(nEH, TB), TB, 0, stream>>>(ei, as_, ad_, m_, E, N);
  edge_expsum  <<<nblk(nEH, TB), TB, 0, stream>>>(ei, as_, ad_, m_, s_, E, N);
  edge_aggregate<<<nblk(nEH, TB), TB, 0, stream>>>(ei, as_, ad_, m_, s_, hfeat, outac, E, N);

  mean_heads_bias<<<nblk((long long)N * DIM_ENC, TB), TB, 0, stream>>>(outac, b1, h1bf, N);

  // ================= Layer 2: GATConv(h1, W2) concat=True ==================
  transpose_cvt_bf16<<<nblk(64 * 512, TB), TB, 0, stream>>>(W2, wbf, 64, 512);
  run_gemm(h1bf, wbf, hfeat, N, DIM_HID, 64, stream);

  node_alpha<<<nblk((long long)N * HEADS, TB), TB, 0, stream>>>(hfeat, asrc2, adst2, as_, ad_, N);
  fill_f32<<<nblk((long long)N * HEADS, TB), TB, 0, stream>>>(m_, -1e30f, N * HEADS);
  fill_f32<<<nblk((long long)N * HEADS, TB), TB, 0, stream>>>(s_, 0.f, N * HEADS);
  fill_f32<<<nblk((long long)N * DIM_HID, TB), TB, 0, stream>>>(outac, 0.f, N * DIM_HID);

  edge_max     <<<nblk(nEH, TB), TB, 0, stream>>>(ei, as_, ad_, m_, E, N);
  edge_expsum  <<<nblk(nEH, TB), TB, 0, stream>>>(ei, as_, ad_, m_, s_, E, N);
  edge_aggregate<<<nblk(nEH, TB), TB, 0, stream>>>(ei, as_, ad_, m_, s_, hfeat, outac, E, N);

  // ================= Pool + MLP ===========================================
  fill_f32<<<nblk((long long)G * DIM_HID, TB), TB, 0, stream>>>(g_, 0.f, G * DIM_HID);
  bias_pool<<<nblk((long long)N * DIM_HID, TB), TB, 0, stream>>>(outac, b2, batch, g_, N);

  cvt_f32_bf16<<<nblk((long long)G * DIM_HID, TB), TB, 0, stream>>>(g_, gbf, G * DIM_HID);
  transpose_cvt_bf16<<<nblk(512 * 512, TB), TB, 0, stream>>>(fcW1, wbf, 512, 512);
  run_gemm(gbf, wbf, g1f, G, DIM_MLP, DIM_HID, stream);
  bias_relu_cvt<<<nblk((long long)G * DIM_MLP, TB), TB, 0, stream>>>(g1f, fcb1, g1bf, G, DIM_MLP);

  transpose_cvt_bf16<<<nblk(512 * 512, TB), TB, 0, stream>>>(fcW2, wbf, 512, 512);
  run_gemm(g1bf, wbf, g2f, G, DIM_MLP, DIM_MLP, stream);
  bias_relu_cvt<<<nblk((long long)G * DIM_MLP, TB), TB, 0, stream>>>(g2f, fcb2, g1bf, G, DIM_MLP);

  final_fc<<<nblk((long long)G * 32, TB), TB, 0, stream>>>(g2f, fcW3, fcb3, out, G);
}
